// DASAttentionGate_47132971106552
// MI455X (gfx1250) — matrix operand order repaired and checked by reading, not compiled, not run on hardware
//
#include <hip/hip_runtime.h>
#include <math.h>

// Problem constants (from reference setup_inputs)
#define BATCH 4
#define CH    128
#define OC    128
#define HH    96
#define WW    96
#define HW    (HH*WW)        // 9216
#define NTOT  (BATCH*HW)     // 36864  (16 | 96, so every 16-wide n-tile stays in one image row)
#define KD    (9*CH)         // 1152  (k = tap*128 + c ordering)
#define EPSV  1e-5f

typedef __attribute__((ext_vector_type(16))) _Float16 v16h;
typedef __attribute__((ext_vector_type(8)))  _Float16 v8h;
typedef __attribute__((ext_vector_type(8)))  float    v8f;

union V16 { v16h v; v8h v8[2]; _Float16 h[16]; };
union V8F { v8f v; float f[8]; };

__device__ __forceinline__ v8f wmma16(v16h a, v16h b, v8f c) {
  // D = A(16x32 f16) * B(32x16 f16) + C(16x16 f32)
  return __builtin_amdgcn_wmma_f32_16x16x32_f16(false, a, false, b, (short)0, c, false, false);
}
__device__ __forceinline__ int iclamp(int v, int lo, int hi) {
  return v < lo ? lo : (v > hi ? hi : v);
}

// ---------------- weight repacks (f32 -> f16, k = tap*128 + c) ----------------
__global__ void k_prep_pw(const float* __restrict__ w, _Float16* __restrict__ a16) {
  int t = blockIdx.x * blockDim.x + threadIdx.x;
  if (t < OC * CH) a16[t] = (_Float16)w[t];            // [o][c] identical layout
}
__global__ void k_prep_off(const float* __restrict__ w, _Float16* __restrict__ a16) {
  int t = blockIdx.x * blockDim.x + threadIdx.x;
  if (t >= 32 * KD) return;                            // M padded 18 -> 32
  int m = t / KD, r = t % KD, tap = r / CH, c = r % CH;
  float val = (m < 18) ? w[(m * CH + c) * 9 + tap] : 0.f;  // off_w [18][128][3][3]
  a16[t] = (_Float16)val;
}
__global__ void k_prep_dc(const float* __restrict__ w, _Float16* __restrict__ a16) {
  int t = blockIdx.x * blockDim.x + threadIdx.x;
  if (t >= OC * KD) return;
  int o = t / KD, r = t % KD, tap = r / CH, c = r % CH;
  a16[t] = (_Float16)w[(o * CH + c) * 9 + tap];        // dc_w [128][128][3][3]
}

// ---------------- depthwise 3x3 (SAME) -> f16 [n][c] ----------------
__global__ void k_dwconv(const float* __restrict__ x, const float* __restrict__ dww,
                         const float* __restrict__ dwb, _Float16* __restrict__ h16) {
  long long t = (long long)blockIdx.x * blockDim.x + threadIdx.x;
  if (t >= (long long)CH * NTOT) return;
  int c = (int)(t / NTOT), n = (int)(t % NTOT);
  int b = n / HW, hw = n % HW, y = hw / WW, xx = hw % WW;
  const float* xp = x + ((size_t)b * CH + c) * HW;
  float acc = dwb[c];
#pragma unroll
  for (int ty = 0; ty < 3; ++ty)
#pragma unroll
    for (int tx = 0; tx < 3; ++tx) {
      int yy = y + ty - 1, xc = xx + tx - 1;
      if (yy >= 0 && yy < HH && xc >= 0 && xc < WW)
        acc += xp[yy * WW + xc] * dww[c * 9 + ty * 3 + tx];
    }
  h16[(size_t)n * CH + c] = (_Float16)acc;
}

// ---------------- pointwise conv as WMMA GEMM: out[o][n] = W[o][c] * h16[n][c] ----------------
// B tile (128k x 16n f16 = 4 KB) staged once per block in LDS; 8 waves cover M=128.
__global__ __launch_bounds__(256) void k_gemm_pw(const _Float16* __restrict__ a16,
                                                 const _Float16* __restrict__ h16,
                                                 const float* __restrict__ bias,
                                                 float* __restrict__ out) {
  __shared__ _Float16 sB[16 * CH];                          // [n_local][c]
  int tid = threadIdx.x;
  int nbase = blockIdx.x * 16;
  {
    int base = tid * 8;                                     // 2048 halves / 256 thr
    int nl = base / CH, c = base % CH;
    *(v8h*)(&sB[nl * CH + c]) = *(const v8h*)(h16 + (size_t)(nbase + nl) * CH + c);
  }
  __syncthreads();

  int lane = tid & 31, wave = tid >> 5;
  int half = lane >> 4, l16 = lane & 15;
  int n = nbase + l16;
  int mbase = wave * 16, m = mbase + l16;
  V8F acc;
#pragma unroll
  for (int v = 0; v < 8; ++v) acc.f[v] = 0.f;
  const _Float16* arow = a16 + (size_t)m * CH;
#pragma unroll
  for (int kb = 0; kb < CH; kb += 32) {
    V16 a, bf;
    a.v     = *(const v16h*)(arow + kb + 16 * half);        // k = kb + 16*half + [0..15]
    bf.v    = *(const v16h*)(&sB[l16 * CH + kb + 16 * half]);  // k = kb + 16*half + [0..15]
    acc.v = wmma16(a.v, bf.v, acc.v);
  }
  int b = n / HW, hw = n % HW;
#pragma unroll
  for (int v = 0; v < 8; ++v) {
    int mo = mbase + v + 8 * half;                          // D layout: M = v + 8*half
    out[((size_t)b * OC + mo) * HW + hw] = acc.f[v] + bias[mo];
  }
}

// ---------------- InstanceNorm: per (b,c) mean/rstd then apply ----------------
__global__ __launch_bounds__(256) void k_inorm_stats(const float* __restrict__ h,
                                                     float* __restrict__ stats) {
  int bc = blockIdx.x;
  const float* p = h + (size_t)bc * HW;
  float s = 0.f, s2 = 0.f;
  for (int i = threadIdx.x; i < HW; i += 256) { float v = p[i]; s += v; s2 += v * v; }
  __shared__ float sh[256], sh2[256];
  sh[threadIdx.x] = s; sh2[threadIdx.x] = s2;
  __syncthreads();
  for (int st = 128; st > 0; st >>= 1) {
    if (threadIdx.x < st) { sh[threadIdx.x] += sh[threadIdx.x + st]; sh2[threadIdx.x] += sh2[threadIdx.x + st]; }
    __syncthreads();
  }
  if (threadIdx.x == 0) {
    float mean = sh[0] / (float)HW;
    float var  = sh2[0] / (float)HW - mean * mean;
    stats[2 * bc]     = mean;
    stats[2 * bc + 1] = rsqrtf(var + EPSV);
  }
}
__global__ void k_inorm_apply(const float* __restrict__ h, const float* __restrict__ stats,
                              float* __restrict__ dsc32, _Float16* __restrict__ dsc16) {
  long long t = (long long)blockIdx.x * blockDim.x + threadIdx.x;
  if (t >= (long long)BATCH * CH * HW) return;
  int hw = (int)(t % HW); int bc = (int)(t / HW); int c = bc % CH; int b = bc / CH;
  float v = (h[t] - stats[2 * bc]) * stats[2 * bc + 1];
  dsc32[t] = v;                                           // BCHW f32 (final gate input)
  dsc16[((size_t)(b * HW + hw)) * CH + c] = (_Float16)v;  // NHWC-ish f16 (GEMM/gather source)
}

// ---------------- offset conv (3x3 SAME, M=18 padded to 32) via WMMA ----------------
__global__ __launch_bounds__(64) void k_gemm_off(const _Float16* __restrict__ a16,
                                                 const _Float16* __restrict__ dsc16,
                                                 const float* __restrict__ bias,
                                                 float* __restrict__ offout) {
  int lane = threadIdx.x & 31, wave = threadIdx.x >> 5;
  int half = lane >> 4, l16 = lane & 15;
  int n = blockIdx.x * 16 + l16;
  int mbase = wave * 16, m = mbase + l16;
  int b = n / HW, hw = n % HW, y = hw / WW, xx = hw % WW;
  V8F acc;
#pragma unroll
  for (int v = 0; v < 8; ++v) acc.f[v] = 0.f;
  const _Float16* arow = a16 + (size_t)m * KD;
  for (int tap = 0; tap < 9; ++tap) {
    int yy = y + tap / 3 - 1, xc = xx + tap % 3 - 1;
    bool valid = (yy >= 0 && yy < HH && xc >= 0 && xc < WW);
    const _Float16* bp = dsc16 +
        ((size_t)b * HW + iclamp(yy, 0, HH - 1) * WW + iclamp(xc, 0, WW - 1)) * CH;
    _Float16 sc = valid ? (_Float16)1.0f : (_Float16)0.0f;  // branchless zero-pad
    const _Float16* atap = arow + tap * CH;
#pragma unroll
    for (int cc = 0; cc < CH; cc += 32) {
      V16 a, bf;
      a.v     = *(const v16h*)(atap + cc + 16 * half);
      bf.v    = *(const v16h*)(bp + cc + 16 * half);
#pragma unroll
      for (int e = 0; e < 16; ++e) bf.h[e] *= sc;
      acc.v = wmma16(a.v, bf.v, acc.v);
    }
  }
#pragma unroll
  for (int v = 0; v < 8; ++v) {
    int mo = mbase + v + 8 * half;
    if (mo < 18) offout[((size_t)b * 18 + mo) * HW + hw] = acc.f[v] + bias[mo];
  }
}

// ---------------- deformable conv: LDS-staged bilinear B tile + WMMA ----------------
// Phase A: 144 threads compute per-(column,tap) corner offsets + masked weights.
// Phase B: 256 threads cooperatively blend the 1152x16 f16 B tile into LDS (36 KB).
// Phase C: 8 waves run the 36-chunk WMMA loop; B fragments come from 32B ds loads.
struct Corner { int o00, o01, o10, o11; float w00, w01, w10, w11; };

__global__ __launch_bounds__(256) void k_gemm_dc(const _Float16* __restrict__ a16,
                                                 const _Float16* __restrict__ dsc16,
                                                 const float* __restrict__ offset,
                                                 const float* __restrict__ bias,
                                                 float* __restrict__ out) {
  __shared__ Corner  sc[16 * 9];                 // 4.6 KB
  __shared__ _Float16 sB[16 * KD];               // 36 KB, [n_local][k]
  int tid = threadIdx.x;
  int nbase = blockIdx.x * 16;
  int b = nbase / HW, hw0 = nbase % HW;          // whole 16-tile shares b and y (16 | 96)
  int y = hw0 / WW, xbase = hw0 % WW;

  // ---- Phase A: corner/weight precompute ----
  if (tid < 144) {
    int nl = tid / 9, tap = tid % 9;
    int xx = xbase + nl, hw = y * WW + xx;
    const float* ob = offset + (size_t)b * 18 * HW + hw;
    float dy = ob[(2 * tap) * HW];
    float dx = ob[(2 * tap + 1) * HW];
    float ys = (float)(y + tap / 3 - 1) + dy;
    float xs = (float)(xx + tap % 3 - 1) + dx;
    float y0f = floorf(ys), x0f = floorf(xs);
    float wy = ys - y0f, wx = xs - x0f;
    int y0 = (int)y0f, x0 = (int)x0f, y1 = y0 + 1, x1 = x0 + 1;
    float vy0 = (y0 >= 0 && y0 < HH) ? 1.f : 0.f;
    float vy1 = (y1 >= 0 && y1 < HH) ? 1.f : 0.f;
    float vx0 = (x0 >= 0 && x0 < WW) ? 1.f : 0.f;
    float vx1 = (x1 >= 0 && x1 < WW) ? 1.f : 0.f;
    Corner k;
    k.w00 = (1.f - wy) * (1.f - wx) * vy0 * vx0;
    k.w01 = (1.f - wy) * wx * vy0 * vx1;
    k.w10 = wy * (1.f - wx) * vy1 * vx0;
    k.w11 = wy * wx * vy1 * vx1;
    int yc0 = iclamp(y0, 0, HH - 1), yc1 = iclamp(y1, 0, HH - 1);
    int xc0 = iclamp(x0, 0, WW - 1), xc1 = iclamp(x1, 0, WW - 1);
    k.o00 = (yc0 * WW + xc0) * CH;
    k.o01 = (yc0 * WW + xc1) * CH;
    k.o10 = (yc1 * WW + xc0) * CH;
    k.o11 = (yc1 * WW + xc1) * CH;
    sc[tid] = k;
  }
  __syncthreads();

  // ---- Phase B: gather + blend B tile (1152 units of 16 channels) ----
  const _Float16* dbase = dsc16 + (size_t)b * HW * CH;
  for (int u = tid; u < 16 * 9 * 8; u += 256) {
    int nl = u / 72, r = u % 72, tap = r / 8, cch = r % 8;
    Corner k = sc[nl * 9 + tap];
    int cb = cch * 16;
    V16 t00, t01, t10, t11, bf;
    t00.v = *(const v16h*)(dbase + k.o00 + cb);
    t01.v = *(const v16h*)(dbase + k.o01 + cb);
    t10.v = *(const v16h*)(dbase + k.o10 + cb);
    t11.v = *(const v16h*)(dbase + k.o11 + cb);
#pragma unroll
    for (int e = 0; e < 16; ++e)
      bf.h[e] = (_Float16)(k.w00 * (float)t00.h[e] + k.w01 * (float)t01.h[e] +
                           k.w10 * (float)t10.h[e] + k.w11 * (float)t11.h[e]);
    *(v16h*)(&sB[nl * KD + tap * CH + cb]) = bf.v;
  }
  __syncthreads();

  // ---- Phase C: WMMA over K = 1152 ----
  int lane = tid & 31, wave = tid >> 5;
  int half = lane >> 4, l16 = lane & 15;
  int mbase = wave * 16, m = mbase + l16;
  V8F acc;
#pragma unroll
  for (int v = 0; v < 8; ++v) acc.f[v] = 0.f;
  const _Float16* arow = a16 + (size_t)m * KD;
  const _Float16* brow = &sB[l16 * KD];
  for (int kb = 0; kb < KD; kb += 32) {
    V16 a, bf;
    a.v     = *(const v16h*)(arow + kb + 16 * half);
    bf.v    = *(const v16h*)(brow + kb + 16 * half);
    acc.v = wmma16(a.v, bf.v, acc.v);
  }
  int hw = hw0 + l16;
#pragma unroll
  for (int v = 0; v < 8; ++v) {
    int mo = mbase + v + 8 * half;
    out[((size_t)b * OC + mo) * HW + hw] = acc.f[v] + bias[mo];
  }
}

// ---------------- channel LayerNorm + sigmoid gate ----------------
__global__ __launch_bounds__(128) void k_final(const float* __restrict__ dc,
                                               const float* __restrict__ dsc32,
                                               const float* __restrict__ ln_g,
                                               const float* __restrict__ ln_b,
                                               float* __restrict__ out) {
  int n = blockIdx.x;                 // b*HW + hw
  int b = n / HW, hw = n % HW;
  int c = threadIdx.x;
  size_t idx = ((size_t)b * OC + c) * HW + hw;
  float v = dc[idx];
  __shared__ float sh[128], sh2[128];
  sh[c] = v; sh2[c] = v * v;
  __syncthreads();
  for (int st = 64; st > 0; st >>= 1) {
    if (c < st) { sh[c] += sh[c + st]; sh2[c] += sh2[c + st]; }
    __syncthreads();
  }
  float mean = sh[0] * (1.f / OC);
  float var  = sh2[0] * (1.f / OC) - mean * mean;
  float r = rsqrtf(var + EPSV);
  float ln = (v - mean) * r * ln_g[c] + ln_b[c];
  float attn = 1.f / (1.f + __expf(-ln));
  out[idx] = dsc32[idx] * attn;
}

extern "C" void kernel_launch(void* const* d_in, const int* in_sizes, int n_in,
                              void* d_out, int out_size, void* d_ws, size_t ws_size,
                              hipStream_t stream) {
  const float* x     = (const float*)d_in[0];
  const float* dw_w  = (const float*)d_in[1];
  const float* dw_b  = (const float*)d_in[2];
  const float* pw_w  = (const float*)d_in[3];
  const float* pw_b  = (const float*)d_in[4];
  const float* off_w = (const float*)d_in[5];
  const float* off_b = (const float*)d_in[6];
  const float* dc_w  = (const float*)d_in[7];
  const float* dc_b  = (const float*)d_in[8];
  const float* ln_g  = (const float*)d_in[9];
  const float* ln_b  = (const float*)d_in[10];
  float* out = (float*)d_out;

  // Workspace carve-up (~60 MB total)
  char* ws = (char*)d_ws;
  size_t o = 0;
  float*    dsc32 = (float*)(ws + o);    o += (size_t)NTOT * CH * 4;   // normalized dsc, BCHW
  float*    pwdc  = (float*)(ws + o);    o += (size_t)NTOT * CH * 4;   // pw output, reused as deform output
  _Float16* dsc16 = (_Float16*)(ws + o); o += (size_t)NTOT * CH * 2;   // dsc f16, [n][c]
  _Float16* h16   = (_Float16*)(ws + o); o += (size_t)NTOT * CH * 2;   // depthwise out f16, [n][c]
  float*    offo  = (float*)(ws + o);    o += (size_t)NTOT * 18 * 4;   // offsets, B x 18 x HW
  _Float16* pwA   = (_Float16*)(ws + o); o += (size_t)OC * CH * 2;     // pw weights f16
  _Float16* offA  = (_Float16*)(ws + o); o += (size_t)32 * KD * 2;     // offset-conv weights f16 (padded)
  _Float16* dcA   = (_Float16*)(ws + o); o += (size_t)OC * KD * 2;     // deform-conv weights f16
  float*    stats = (float*)(ws + o);    o += (size_t)BATCH * CH * 2 * 4;
  (void)ws_size; (void)in_sizes; (void)n_in; (void)out_size;

  k_prep_pw <<<(OC * CH + 255) / 256, 256, 0, stream>>>(pw_w, pwA);
  k_prep_off<<<(32 * KD + 255) / 256, 256, 0, stream>>>(off_w, offA);
  k_prep_dc <<<(OC * KD + 255) / 256, 256, 0, stream>>>(dc_w, dcA);

  k_dwconv<<<(int)(((long long)CH * NTOT + 255) / 256), 256, 0, stream>>>(x, dw_w, dw_b, h16);

  k_gemm_pw<<<NTOT / 16, 256, 0, stream>>>(pwA, h16, pw_b, pwdc);

  k_inorm_stats<<<BATCH * CH, 256, 0, stream>>>(pwdc, stats);
  k_inorm_apply<<<(int)(((long long)BATCH * CH * HW + 255) / 256), 256, 0, stream>>>(pwdc, stats, dsc32, dsc16);

  k_gemm_off<<<NTOT / 16, 64, 0, stream>>>(offA, dsc16, off_b, offo);

  k_gemm_dc<<<NTOT / 16, 256, 0, stream>>>(dcA, dsc16, offo, dc_b, pwdc);

  k_final<<<NTOT, 128, 0, stream>>>(pwdc, dsc32, ln_g, ln_b, out);
}
